// MoE_72456098283872
// MI455X (gfx1250) — compile-verified, matrix operand order checked by reading, not run on hardware
//
#include <hip/hip_runtime.h>
#include <hip/hip_bf16.h>

typedef __attribute__((ext_vector_type(16))) __bf16 v16bf;
typedef __attribute__((ext_vector_type(8)))  __bf16 bf16x8;
typedef __attribute__((ext_vector_type(4)))  __bf16 bf16x4;
typedef __attribute__((ext_vector_type(2)))  __bf16 bf16x2;
typedef __attribute__((ext_vector_type(8)))  float  v8f;

#define B_TOK   8192
#define D_IN    1024
#define D_HID   2048
#define D_OUT   1024
#define NEXP    8
#define NMET    3
#define TOPK    4
#define LOSS_COEF 0.01f
#define EPS_F   2.2204460492503131e-16f   // np.finfo(float64).eps, as in reference

// Build a v16bf WMMA fragment from a row-major LDS row (k contiguous).
// ISA 7.12.2 16-bit A/B layout: element j -> k = koff + (lane&16)>>1 + ((j&8)<<1) + (j&7)
// => two contiguous 8-element (16 B) runs at koff+off and koff+off+16.
__device__ __forceinline__ v16bf frag_from_row(const __bf16* row, int lane, int ksub) {
    v16bf f;
    const __bf16* p = row + (ksub << 5) + ((lane & 16) >> 1);
    #pragma unroll
    for (int j = 0; j < 8; ++j) f[j] = p[j];
    #pragma unroll
    for (int j = 0; j < 8; ++j) f[8 + j] = p[16 + j];
    return f;
}

// ---------------------------------------------------------------------------
// Gating: logits = x @ w_gate[g]; top-4 softmax -> gates; per-block
// importance/load partials in fixed order (deterministic). One wave per token.
// ---------------------------------------------------------------------------
__global__ __launch_bounds__(256) void moe_gating(
    const float* __restrict__ x, const float* __restrict__ wgate,
    float* __restrict__ gates, float* __restrict__ impPart, float* __restrict__ loadPart)
{
    __shared__ float xs[8][D_IN];                 // 32 KB
    __shared__ float wgs[NMET * NEXP][D_IN];      // 96 KB, transposed for bank-free dots
    __shared__ float lgv[8][NMET * NEXP];
    __shared__ float lld[8][NMET * NEXP];

    const int tid = threadIdx.x;
    const int lane = tid & 31, wid = tid >> 5;
    const int b0 = blockIdx.x * 8;

    #pragma unroll
    for (int i = 0; i < 32; ++i) {                // 8 rows * 1024
        int idx = i * 256 + tid;
        xs[idx >> 10][idx & 1023] = x[(size_t)(b0 + (idx >> 10)) * D_IN + (idx & 1023)];
    }
    #pragma unroll
    for (int i = 0; i < 96; ++i) {                // 3*1024*8 = 24576 elements
        int idx = i * 256 + tid;                  // wgate flat: (g*1024 + i)*8 + e
        int g = idx >> 13;
        int rest = idx & 8191;
        wgs[g * NEXP + (rest & 7)][rest >> 3] = wgate[idx];
    }
    __syncthreads();

    const int b = b0 + wid;
    #pragma unroll
    for (int g = 0; g < NMET; ++g) {
        float logit[NEXP];
        #pragma unroll
        for (int e = 0; e < NEXP; ++e) {
            float p = 0.f;
            #pragma unroll
            for (int i = lane; i < D_IN; i += 32)
                p += xs[wid][i] * wgs[g * NEXP + e][i];
            #pragma unroll
            for (int off = 16; off > 0; off >>= 1) p += __shfl_xor(p, off);
            logit[e] = p;                          // all lanes hold full sum
        }
        // top-4 (replicated in every lane)
        int used = 0;
        float sv[TOPK]; int si[TOPK];
        #pragma unroll
        for (int t = 0; t < TOPK; ++t) {
            float best = -3.4e38f; int bi = 0;
            #pragma unroll
            for (int e = 0; e < NEXP; ++e)
                if (!((used >> e) & 1) && logit[e] > best) { best = logit[e]; bi = e; }
            used |= 1 << bi; sv[t] = best; si[t] = bi;
        }
        float s = 0.f, ex[TOPK];
        #pragma unroll
        for (int t = 0; t < TOPK; ++t) { ex[t] = __expf(sv[t] - sv[0]); s += ex[t]; }
        const float inv = 1.f / s;
        if (lane == 0) {
            #pragma unroll
            for (int e = 0; e < NEXP; ++e) {
                float gv = 0.f;
                #pragma unroll
                for (int t = 0; t < TOPK; ++t) if (si[t] == e) gv = ex[t] * inv;
                gates[((size_t)g * B_TOK + b) * NEXP + e] = gv;
                lgv[wid][g * NEXP + e] = gv;
                lld[wid][g * NEXP + e] = (gv > 0.f) ? 1.f : 0.f;
            }
        }
    }
    __syncthreads();
    if (tid < NMET * NEXP) {                       // ordered wave sum => deterministic
        float a = 0.f, c = 0.f;
        #pragma unroll
        for (int w = 0; w < 8; ++w) { a += lgv[w][tid]; c += lld[w][tid]; }
        impPart [(size_t)blockIdx.x * (NMET * NEXP) + tid] = a;
        loadPart[(size_t)blockIdx.x * (NMET * NEXP) + tid] = c;
    }
}

// ---------------------------------------------------------------------------
// CV loss: var(ddof=1)/(mean^2 + 1e-8), summed over metrics * 0.01
// ---------------------------------------------------------------------------
__device__ __forceinline__ float cv8(const float* v) {
    float m = 0.f;
    #pragma unroll
    for (int e = 0; e < NEXP; ++e) m += v[e];
    m *= 0.125f;
    float s = 0.f;
    #pragma unroll
    for (int e = 0; e < NEXP; ++e) { float d = v[e] - m; s += d * d; }
    return (s * (1.f / 7.f)) / (m * m + 1e-8f);
}

__global__ void moe_loss(const float* __restrict__ impPart, const float* __restrict__ loadPart,
                         float* __restrict__ lossOut)
{
    __shared__ float simp[NMET * NEXP], sload[NMET * NEXP];
    const int t = threadIdx.x;
    if (t < NMET * NEXP) {
        float a = 0.f, c = 0.f;
        for (int bkt = 0; bkt < B_TOK / 8; ++bkt) {    // fixed order reduction
            a += impPart [(size_t)bkt * (NMET * NEXP) + t];
            c += loadPart[(size_t)bkt * (NMET * NEXP) + t];
        }
        simp[t] = a; sload[t] = c;
    }
    __syncthreads();
    if (t == 0) {
        float total = 0.f;
        #pragma unroll
        for (int g = 0; g < NMET; ++g)
            total += LOSS_COEF * (cv8(&simp[g * NEXP]) + cv8(&sload[g * NEXP]));
        lossOut[0] = total;
    }
}

// ---------------------------------------------------------------------------
// Pass A: H[e] = relu(x @ w1[e] + b1[e])  (bf16 WMMA, bf16 output to ws)
// 64x128 tile / 256-thread WG (8 waves as 4Mx2N, 16x64 per wave, 4 accums).
// K-step 64. W1 staged TRANSPOSED [n][k] with pair-packed b32 stores.
// ---------------------------------------------------------------------------
__global__ __launch_bounds__(256) void moe_h_pass(
    const float* __restrict__ x, const float* __restrict__ w1, const float* __restrict__ b1,
    __bf16* __restrict__ H, int row0, int bseg)
{
    __shared__ __bf16 Xs[64][72];      // [m][k]  row stride 144 B (16B-aligned runs)
    __shared__ __bf16 WT[128][72];     // [n][k]  transposed
    const int e     = blockIdx.z;
    const int mbase = blockIdx.x * 64;     // segment-local row
    const int nbase = blockIdx.y * 128;
    const int tid = threadIdx.x;
    const int lane = tid & 31, wid = tid >> 5;
    const int waveM = wid >> 1, waveN = wid & 1;
    const int mhalf = (lane >> 4) & 1;
    const int nloc  = lane & 15;
    const float* w1e = w1 + (size_t)e * D_IN * D_HID;

    const v8f vzero = {0,0,0,0,0,0,0,0};
    v8f c[4];
    #pragma unroll
    for (int nt = 0; nt < 4; ++nt) c[nt] = vzero;

    for (int k0 = 0; k0 < D_IN; k0 += 64) {
        // X tile 64x64 (fp32 -> bf16): float4 read -> one bf16x4 (8 B) LDS store
        #pragma unroll
        for (int i = 0; i < 4; ++i) {
            int idx4 = i * 256 + tid;              // 1024 float4s
            int r = idx4 >> 4, c4 = (idx4 & 15) << 2;
            float4 v = *reinterpret_cast<const float4*>(
                &x[(size_t)(row0 + mbase + r) * D_IN + k0 + c4]);
            bf16x4 p = {(__bf16)v.x, (__bf16)v.y, (__bf16)v.z, (__bf16)v.w};
            *reinterpret_cast<bf16x4*>(&Xs[r][c4]) = p;
        }
        // W1 tile 64(k)x128(n): two float4s from consecutive k rows ->
        // pair-packed bf16x2 stores (adjacent k in transposed layout)
        #pragma unroll
        for (int i = 0; i < 4; ++i) {
            int idx = i * 256 + tid;               // 1024 pair-units (32 rp x 32 c4)
            int r = (idx >> 5) << 1, c4 = (idx & 31) << 2;
            float4 va = *reinterpret_cast<const float4*>(
                &w1e[(size_t)(k0 + r) * D_HID + nbase + c4]);
            float4 vb = *reinterpret_cast<const float4*>(
                &w1e[(size_t)(k0 + r + 1) * D_HID + nbase + c4]);
            bf16x2 p0 = {(__bf16)va.x, (__bf16)vb.x};
            bf16x2 p1 = {(__bf16)va.y, (__bf16)vb.y};
            bf16x2 p2 = {(__bf16)va.z, (__bf16)vb.z};
            bf16x2 p3 = {(__bf16)va.w, (__bf16)vb.w};
            *reinterpret_cast<bf16x2*>(&WT[c4 + 0][r]) = p0;
            *reinterpret_cast<bf16x2*>(&WT[c4 + 1][r]) = p1;
            *reinterpret_cast<bf16x2*>(&WT[c4 + 2][r]) = p2;
            *reinterpret_cast<bf16x2*>(&WT[c4 + 3][r]) = p3;
        }
        __syncthreads();

        const int m = (waveM << 4) + nloc;
        #pragma unroll
        for (int ks = 0; ks < 2; ++ks) {
            v16bf a = frag_from_row(&Xs[m][0], lane, ks);
            #pragma unroll
            for (int nt = 0; nt < 4; ++nt) {
                const int n = (waveN << 6) + (nt << 4) + nloc;
                v16bf bfrag = frag_from_row(&WT[n][0], lane, ks);
                c[nt] = __builtin_amdgcn_wmma_f32_16x16x32_bf16(false, a, false, bfrag,
                                                                (short)0, c[nt], false, false);
            }
        }
        __syncthreads();
    }

    #pragma unroll
    for (int nt = 0; nt < 4; ++nt) {
        #pragma unroll
        for (int r = 0; r < 8; ++r) {
            const int Ml  = mbase + (waveM << 4) + mhalf * 8 + r;
            const int col = nbase + (waveN << 6) + (nt << 4) + nloc;
            float v = c[nt][r] + b1[(size_t)e * D_HID + col];
            v = v > 0.f ? v : 0.f;
            H[((size_t)e * bseg + Ml) * D_HID + col] = (__bf16)v;
        }
    }
}

// ---------------------------------------------------------------------------
// Pass B: for each (32-row x 256-col) output rectangle, loop experts:
//   out_e = H_e @ w2_e + b2_e ; comb[g] += gate[g,b,e] * exp(out_e)
// then write log(comb or eps). 8 waves as 2Mx4N, 16x64 per wave (4 accums).
// K-step 64. W2 staged TRANSPOSED [n][k] with pair-packed b32 stores.
// ---------------------------------------------------------------------------
__global__ __launch_bounds__(256) void moe_combine(
    const __bf16* __restrict__ H, const float* __restrict__ w2, const float* __restrict__ b2,
    const float* __restrict__ gates, float* __restrict__ out,
    int row0, int bseg)
{
    __shared__ __bf16 Hs[32][72];      // [m][k]
    __shared__ __bf16 WT[256][72];     // [n][k]  transposed  (36.9 KB)
    __shared__ float  gl[NMET][32];
    const int rbase = blockIdx.x * 32;     // segment-local row
    const int obase = blockIdx.y * 256;
    const int tid = threadIdx.x;
    const int lane = tid & 31, wid = tid >> 5;
    const int waveM = wid >> 2, waveN = wid & 3;
    const int mhalf = (lane >> 4) & 1;
    const int nloc  = lane & 15;

    float comb[NMET][4][8];
    #pragma unroll
    for (int g = 0; g < NMET; ++g)
        #pragma unroll
        for (int nt = 0; nt < 4; ++nt)
            #pragma unroll
            for (int r = 0; r < 8; ++r) comb[g][nt][r] = 0.f;

    const v8f vzero = {0,0,0,0,0,0,0,0};

    for (int e = 0; e < NEXP; ++e) {
        if (tid < NMET * 32) {
            int g = tid >> 5, r = tid & 31;
            gl[g][r] = gates[((size_t)g * B_TOK + (row0 + rbase + r)) * NEXP + e];
        }
        v8f c[4];
        #pragma unroll
        for (int nt = 0; nt < 4; ++nt) c[nt] = vzero;
        const float* w2e = w2 + (size_t)e * D_HID * D_OUT;
        __syncthreads();

        for (int k0 = 0; k0 < D_HID; k0 += 64) {
            // H tile 32x64 bf16: one 16 B (8-elem) vector per thread
            {
                int r = tid >> 3, c8 = (tid & 7) << 3;
                bf16x8 v = *reinterpret_cast<const bf16x8*>(
                    &H[((size_t)e * bseg + rbase + r) * D_HID + k0 + c8]);
                *reinterpret_cast<bf16x8*>(&Hs[r][c8]) = v;
            }
            // W2 tile 64(k)x256(n): paired float4 reads -> packed bf16x2 stores
            #pragma unroll
            for (int i = 0; i < 8; ++i) {
                int idx = i * 256 + tid;           // 2048 pair-units (32 rp x 64 c4)
                int r = (idx >> 6) << 1, c4 = (idx & 63) << 2;
                float4 va = *reinterpret_cast<const float4*>(
                    &w2e[(size_t)(k0 + r) * D_OUT + obase + c4]);
                float4 vb = *reinterpret_cast<const float4*>(
                    &w2e[(size_t)(k0 + r + 1) * D_OUT + obase + c4]);
                bf16x2 p0 = {(__bf16)va.x, (__bf16)vb.x};
                bf16x2 p1 = {(__bf16)va.y, (__bf16)vb.y};
                bf16x2 p2 = {(__bf16)va.z, (__bf16)vb.z};
                bf16x2 p3 = {(__bf16)va.w, (__bf16)vb.w};
                *reinterpret_cast<bf16x2*>(&WT[c4 + 0][r]) = p0;
                *reinterpret_cast<bf16x2*>(&WT[c4 + 1][r]) = p1;
                *reinterpret_cast<bf16x2*>(&WT[c4 + 2][r]) = p2;
                *reinterpret_cast<bf16x2*>(&WT[c4 + 3][r]) = p3;
            }
            __syncthreads();

            const int m = (waveM << 4) + nloc;
            #pragma unroll
            for (int ks = 0; ks < 2; ++ks) {
                v16bf a = frag_from_row(&Hs[m][0], lane, ks);
                #pragma unroll
                for (int nt = 0; nt < 4; ++nt) {
                    const int n = (waveN << 6) + (nt << 4) + nloc;
                    v16bf bfrag = frag_from_row(&WT[n][0], lane, ks);
                    c[nt] = __builtin_amdgcn_wmma_f32_16x16x32_bf16(false, a, false, bfrag,
                                                                    (short)0, c[nt], false, false);
                }
            }
            __syncthreads();
        }
        // epilogue: exp + gated accumulation into registers
        #pragma unroll
        for (int nt = 0; nt < 4; ++nt) {
            const int col = obase + (waveN << 6) + (nt << 4) + nloc;
            const float bias = b2[(size_t)e * D_OUT + col];
            #pragma unroll
            for (int r = 0; r < 8; ++r) {
                const float ev = __expf(c[nt][r] + bias);
                const int mrow = (waveM << 4) + mhalf * 8 + r;
                #pragma unroll
                for (int g = 0; g < NMET; ++g) comb[g][nt][r] += gl[g][mrow] * ev;
            }
        }
        __syncthreads();   // protect gl before next expert overwrites it
    }

    #pragma unroll
    for (int nt = 0; nt < 4; ++nt) {
        #pragma unroll
        for (int r = 0; r < 8; ++r) {
            const int grow = row0 + rbase + (waveM << 4) + mhalf * 8 + r;
            const int col  = obase + (waveN << 6) + (nt << 4) + nloc;
            #pragma unroll
            for (int g = 0; g < NMET; ++g) {
                float v = comb[g][nt][r];
                v = (v == 0.f) ? EPS_F : v;
                out[((size_t)g * B_TOK + grow) * D_OUT + col] = __logf(v);
            }
        }
    }
}

// ---------------------------------------------------------------------------
extern "C" void kernel_launch(void* const* d_in, const int* in_sizes, int n_in,
                              void* d_out, int out_size, void* d_ws, size_t ws_size,
                              hipStream_t stream) {
    const float* x  = (const float*)d_in[0];
    const float* wg = (const float*)d_in[1];
    const float* w1 = (const float*)d_in[2];
    const float* b1 = (const float*)d_in[3];
    const float* w2 = (const float*)d_in[4];
    const float* b2 = (const float*)d_in[5];
    float* out = (float*)d_out;

    char*  ws       = (char*)d_ws;
    float* gates    = (float*)ws;                          // 786,432 B
    float* impPart  = (float*)(ws + 786432);               //  98,304 B
    float* loadPart = (float*)(ws + 786432 + 98304);       //  98,304 B  (ends 983,040)
    __bf16* Hbuf    = (__bf16*)(ws + 1048576);             // H: [E][bseg][D_HID] bf16

    // Segment B so the bf16 H buffer fits whatever workspace we got.
    size_t avail    = ws_size > 1048576 ? ws_size - 1048576 : 0;
    size_t rowbytes = (size_t)NEXP * D_HID * sizeof(__bf16);   // 32768 B per token
    long long rows  = (long long)(avail / rowbytes);
    int bseg = rows > B_TOK ? B_TOK : (int)rows;
    bseg &= ~63;                    // multiple of 64 => exact tiling, no guards
    if (bseg < 64) bseg = 64;       // ~3 MB minimum workspace assumption

    moe_gating<<<dim3(B_TOK / 8), 256, 0, stream>>>(x, wg, gates, impPart, loadPart);
    moe_loss<<<1, 32, 0, stream>>>(impPart, loadPart, out + (size_t)NMET * B_TOK * D_OUT);

    for (int row0 = 0; row0 < B_TOK; row0 += bseg) {
        int nrows = B_TOK - row0; if (nrows > bseg) nrows = bseg;
        dim3 gA(nrows / 64, D_HID / 128, NEXP);
        moe_h_pass<<<gA, 256, 0, stream>>>(x, w1, b1, Hbuf, row0, bseg);
        dim3 gB(nrows / 32, D_OUT / 256);
        moe_combine<<<gB, 256, 0, stream>>>(Hbuf, w2, b2, gates, out, row0, bseg);
    }
}
